// CaduceusMixerModel_15358803050512
// MI455X (gfx1250) — matrix operand order repaired and checked by reading, not compiled, not run on hardware
//
#include <hip/hip_runtime.h>
#include <hip/hip_bf16.h>

// ---------------- model constants (from reference) ----------------
#define L_SEQ   4096
#define DM      256    // d_model
#define DI      512    // d_inner
#define DS      16     // d_state
#define DTR     16     // dt_rank
#define DC      4      // d_conv
#define NLAYER  4
#define XDBW    (DTR + 2*DS)   // 48
#define EPSF    1e-5f

typedef __bf16 bf16_t;
typedef __attribute__((ext_vector_type(16))) __bf16 v16bf;
typedef __attribute__((ext_vector_type(8)))  float  v8f;

union AFrag { v16bf v; bf16_t e[16]; };
union CFrag { v8f  v; float  e[8]; };

// ---------------- embedding gather ----------------
__global__ __launch_bounds__(256) void k_embed(const int* __restrict__ ids,
                                               const float* __restrict__ ew,
                                               float* __restrict__ h) {
  int gid = blockIdx.x * 256 + threadIdx.x;           // L*DM threads
  int t = gid >> 8, d = gid & (DM - 1);
  h[(size_t)t * DM + d] = ew[(size_t)ids[t] * DM + d];
}

// ---------------- f32 -> bf16 convert (4 elems/thread) ----------------
__global__ __launch_bounds__(256) void k_cvt(const float* __restrict__ src,
                                             bf16_t* __restrict__ dst, int n) {
  int i = (blockIdx.x * 256 + threadIdx.x) * 4;       // n is a multiple of 4
  if (i < n) {
    float4 f = *(const float4*)(src + i);
    dst[i + 0] = (bf16_t)f.x;
    dst[i + 1] = (bf16_t)f.y;
    dst[i + 2] = (bf16_t)f.z;
    dst[i + 3] = (bf16_t)f.w;
  }
}

// ---------------- fused residual add + RMSNorm ----------------
// resid = first ? x : x + resid ; out = rmsnorm(resid) * w
__global__ __launch_bounds__(256) void k_rmsnorm(const float* __restrict__ x,
                                                 float* __restrict__ resid,
                                                 const float* __restrict__ w,
                                                 bf16_t* __restrict__ obf,
                                                 float* __restrict__ of32,
                                                 int first) {
  __shared__ float sm[8];
  const int t = blockIdx.x, d = threadIdx.x;
  float v = x[(size_t)t * DM + d];
  if (!first) v += resid[(size_t)t * DM + d];
  resid[(size_t)t * DM + d] = v;
  float s = v * v;
  #pragma unroll
  for (int m = 16; m >= 1; m >>= 1) s += __shfl_xor(s, m, 32);
  if ((d & 31) == 0) sm[d >> 5] = s;
  __syncthreads();
  float tot = 0.f;
  #pragma unroll
  for (int wv = 0; wv < 8; ++wv) tot += sm[wv];
  float scale = rsqrtf(tot * (1.f / DM) + EPSF);
  float o = v * scale * w[d];
  if (obf)  obf[(size_t)t * DM + d] = (bf16_t)o;
  if (of32) of32[(size_t)t * DM + d] = o;
}

// ---------------- bf16 WMMA GEMM: C[MxN] (+)= A[MxK] * W[NxK]^T ----------------
// Block = 8 waves -> 64x64 output tile; wave (mi,ni) owns a 16x32 strip
// (one A fragment reused across two v_wmma issues). Tiles are staged through
// double-buffered LDS with one barrier per K-step; global loads are 128-bit.
// Requires: M % 64 == 0, K % 32 == 0, rows 16B-aligned (true for this model).
__global__ __launch_bounds__(256) void k_gemm_bf16(const bf16_t* __restrict__ A,
                                                   const bf16_t* __restrict__ W,
                                                   float* __restrict__ C,
                                                   int M, int N, int K, int acc) {
  __shared__ __align__(16) bf16_t As[2][64][40];   // padded rows (80B pitch)
  __shared__ __align__(16) bf16_t Bs[2][64][40];
  const int tid  = threadIdx.x;
  const int lane = tid & 31, wave = tid >> 5;
  const int mi = wave & 3, ni = wave >> 2;         // 4 x 2 wave grid
  const int half = lane >> 4, l15 = lane & 15;
  const int m0 = blockIdx.x * 64, n0 = blockIdx.y * 64;
  const int lr = tid >> 2;                          // 0..63 tile row
  const int lc = (tid & 3) * 8;                     // 0,8,16,24 tile col

  const bf16_t* Arow = A + (size_t)(m0 + lr) * K + lc;
  const bf16_t* Wrow = W + (size_t)(n0 + lr) * K + lc;
  const bool wvalid = (n0 + lr) < N;

  CFrag c0, c1;
  #pragma unroll
  for (int r = 0; r < 8; ++r) { c0.e[r] = 0.f; c1.e[r] = 0.f; }

  // preload k-slab 0
  uint4 va = *(const uint4*)(Arow);
  uint4 vb = make_uint4(0u, 0u, 0u, 0u);
  if (wvalid) vb = *(const uint4*)(Wrow);
  int buf = 0;
  *(uint4*)(&As[buf][lr][lc]) = va;
  *(uint4*)(&Bs[buf][lr][lc]) = vb;
  __syncthreads();

  for (int kk = 0; kk < K; kk += 32) {
    const bool more = (kk + 32) < K;
    uint4 na = make_uint4(0u, 0u, 0u, 0u);
    uint4 nb = make_uint4(0u, 0u, 0u, 0u);
    if (more) {
      na = *(const uint4*)(Arow + kk + 32);
      if (wvalid) nb = *(const uint4*)(Wrow + kk + 32);
    }

    AFrag a, b0, b1;
    #pragma unroll
    for (int j = 0; j < 16; ++j) {        // A layout: lanes 0-15 K0-7/K16-23, 16-31 K8-15/K24-31
      int ka = j + (j >= 8 ? 8 : 0) + 8 * half;
      a.e[j] = As[buf][16 * mi + l15][ka];
    }
    #pragma unroll
    for (int j = 0; j < 16; ++j) {        // B layout: lanes 0-15 K0-7/K16-23, 16-31 K8-15/K24-31
      int kb = j + (j >= 8 ? 8 : 0) + 8 * half;
      b0.e[j] = Bs[buf][32 * ni + l15][kb];
      b1.e[j] = Bs[buf][32 * ni + 16 + l15][kb];
    }
    c0.v = __builtin_amdgcn_wmma_f32_16x16x32_bf16(false, a.v, false, b0.v,
                                                   (short)0, c0.v, false, false);
    c1.v = __builtin_amdgcn_wmma_f32_16x16x32_bf16(false, a.v, false, b1.v,
                                                   (short)0, c1.v, false, false);
    if (more) {
      *(uint4*)(&As[buf ^ 1][lr][lc]) = na;
      *(uint4*)(&Bs[buf ^ 1][lr][lc]) = nb;
      __syncthreads();                    // uniform: kk,K are block-uniform
      buf ^= 1;
    }
  }

  #pragma unroll
  for (int r = 0; r < 8; ++r) {           // D: VGPR r -> M = r + 8*half
    int row = m0 + 16 * mi + r + 8 * half;
    int col0 = n0 + 32 * ni + l15;
    int col1 = col0 + 16;
    size_t i0 = (size_t)row * N + col0;
    size_t i1 = (size_t)row * N + col1;
    if (col0 < N) C[i0] = acc ? (C[i0] + c0.e[r]) : c0.e[r];
    if (col1 < N) C[i1] = acc ? (C[i1] + c1.e[r]) : c1.e[r];
  }
}

// ---------------- causal depthwise conv1d + SiLU (per direction) ----------------
__global__ __launch_bounds__(256) void k_conv(const float* __restrict__ xz,
                                              const float* __restrict__ cw,
                                              const float* __restrict__ cb,
                                              float* __restrict__ u,
                                              bf16_t* __restrict__ ubf, int dir) {
  int gid = blockIdx.x * 256 + threadIdx.x;     // L*DI threads
  int t = gid >> 9, d = gid & (DI - 1);
  float acc = cb[d];
  #pragma unroll
  for (int j = 0; j < DC; ++j) {
    int tt = t - (DC - 1) + j;
    if (tt >= 0) {
      int row = dir ? (L_SEQ - 1 - tt) : tt;
      acc += cw[d * DC + j] * xz[(size_t)row * (2 * DI) + d];
    }
  }
  float s = acc * (1.f / (1.f + __expf(-acc)));  // SiLU
  u[(size_t)t * DI + d]   = s;
  ubf[(size_t)t * DI + d] = (bf16_t)s;
}

// ---------------- dt = softplus(dt_r @ dtw^T + dtb)  (K=16: f32 VALU) ----------------
__global__ __launch_bounds__(256) void k_dt(const float* __restrict__ xdb,
                                            const float* __restrict__ dtw,
                                            const float* __restrict__ dtb,
                                            float* __restrict__ dt) {
  int gid = blockIdx.x * 256 + threadIdx.x;     // L*DI threads
  int t = gid >> 9, d = gid & (DI - 1);
  float s = dtb[d];
  #pragma unroll
  for (int r = 0; r < DTR; ++r)
    s += xdb[(size_t)t * XDBW + r] * dtw[d * DTR + r];
  s = (s > 20.f) ? s : log1pf(__expf(s));
  dt[(size_t)t * DI + d] = s;
}

// ---------------- selective scan, both directions concurrently ----------------
// Lane (n, ch-sub) owns state h[ch][n]; 16-lane group reduces h*C via shfl_xor.
// y (incl. D*u skip term) is written back in place of u.
__global__ __launch_bounds__(256) void k_scan(float* __restrict__ u0, float* __restrict__ u1,
                                              const float* __restrict__ dt0, const float* __restrict__ dt1,
                                              const float* __restrict__ xdb0, const float* __restrict__ xdb1,
                                              const float* __restrict__ Alog,
                                              const float* __restrict__ Dpv) {
  const int dir = blockIdx.y;
  float* u         = dir ? u1 : u0;
  const float* dt  = dir ? dt1 : dt0;
  const float* xdb = dir ? xdb1 : xdb0;
  const int lane = threadIdx.x & 31, wave = threadIdx.x >> 5;
  const int n  = lane & 15;
  const int ch = (blockIdx.x * 8 + wave) * 2 + (lane >> 4);
  const float Ac = -__expf(Alog[(size_t)dir * DI * DS + ch * DS + n]);
  const float Dv = Dpv[dir * DI + ch];
  float h = 0.f;
  for (int t = 0; t < L_SEQ; ++t) {
    if ((t & 127) == 0 && t + 128 < L_SEQ) {   // hide latency of serial loop
      __builtin_prefetch(&dt[(size_t)(t + 128) * DI + ch], 0, 1);
      __builtin_prefetch(&u[(size_t)(t + 128) * DI + ch], 0, 1);
    }
    const float dtv = dt[(size_t)t * DI + ch];
    const float uv  = u[(size_t)t * DI + ch];
    const float Bv  = xdb[(size_t)t * XDBW + DTR + n];
    const float Cv  = xdb[(size_t)t * XDBW + DTR + DS + n];
    h = __expf(dtv * Ac) * h + (dtv * Bv) * uv;
    float p = h * Cv;
    p += __shfl_xor(p, 8, 32);
    p += __shfl_xor(p, 4, 32);
    p += __shfl_xor(p, 2, 32);
    p += __shfl_xor(p, 1, 32);
    if (n == 0) u[(size_t)t * DI + ch] = p + uv * Dv;   // all lanes read uv first
  }
}

// ---------------- gate: g[pos] = y[t] * silu(z[pos]); reverse un-flips rows ----------------
__global__ __launch_bounds__(256) void k_gate(const float* __restrict__ y,
                                              const float* __restrict__ xz,
                                              bf16_t* __restrict__ gbf, int dir) {
  int gid = blockIdx.x * 256 + threadIdx.x;     // L*DI threads
  int t = gid >> 9, d = gid & (DI - 1);
  int pos = dir ? (L_SEQ - 1 - t) : t;
  float z = xz[(size_t)pos * (2 * DI) + DI + d];
  float g = y[(size_t)t * DI + d] * z * (1.f / (1.f + __expf(-z)));
  gbf[(size_t)pos * DI + d] = (bf16_t)g;
}

// ---------------- host orchestration ----------------
extern "C" void kernel_launch(void* const* d_in, const int* in_sizes, int n_in,
                              void* d_out, int out_size, void* d_ws, size_t ws_size,
                              hipStream_t stream) {
  (void)in_sizes; (void)n_in; (void)out_size; (void)ws_size;
  const int*   ids     = (const int*)  d_in[0];
  const float* embed_w = (const float*)d_in[1];
  const float* norm_w  = (const float*)d_in[2];
  const float* in_w    = (const float*)d_in[3];   // (4,1024,256)
  const float* out_w   = (const float*)d_in[4];   // (4,256,512)
  const float* conv_w  = (const float*)d_in[5];   // (4,2,512,4)
  const float* conv_b  = (const float*)d_in[6];   // (4,2,512)
  const float* xp_w    = (const float*)d_in[7];   // (4,2,48,512)
  const float* dt_w    = (const float*)d_in[8];   // (4,2,512,16)
  const float* dt_b    = (const float*)d_in[9];   // (4,2,512)
  const float* A_log   = (const float*)d_in[10];  // (4,2,512,16)
  const float* Dp      = (const float*)d_in[11];  // (4,2,512)
  const float* normf_w = (const float*)d_in[12];  // (256,)
  float* out = (float*)d_out;

  char* ws = (char*)d_ws;
  size_t off = 0;
  auto alloc = [&](size_t bytes) -> void* {
    void* p = ws + off;
    off = (off + bytes + 255) & ~(size_t)255;
    return p;
  };
  float*  b_resid = (float*) alloc((size_t)L_SEQ * DM * 4);
  float*  b_h     = (float*) alloc((size_t)L_SEQ * DM * 4);
  bf16_t* b_hnbf  = (bf16_t*)alloc((size_t)L_SEQ * DM * 2);
  float*  b_xz    = (float*) alloc((size_t)L_SEQ * 2 * DI * 4);
  float*  b_u[2]; bf16_t* b_ubf[2]; float* b_xdb[2]; float* b_dt[2];
  for (int d = 0; d < 2; ++d) {
    b_u[d]   = (float*) alloc((size_t)L_SEQ * DI * 4);
    b_ubf[d] = (bf16_t*)alloc((size_t)L_SEQ * DI * 2);   // reused for gated output
    b_xdb[d] = (float*) alloc((size_t)L_SEQ * XDBW * 4);
    b_dt[d]  = (float*) alloc((size_t)L_SEQ * DI * 4);
  }
  bf16_t* w_in_bf  = (bf16_t*)alloc((size_t)2 * DI * DM * 2);
  bf16_t* w_x_bf   = (bf16_t*)alloc((size_t)XDBW * DI * 2);
  bf16_t* w_out_bf = (bf16_t*)alloc((size_t)DM * DI * 2);

  const int T256 = 256;
  k_embed<<<(L_SEQ * DM) / T256, T256, 0, stream>>>(ids, embed_w, b_h);

  for (int i = 0; i < NLAYER; ++i) {
    // resid += h ; hn = rmsnorm(resid) (bf16 for WMMA)
    k_rmsnorm<<<L_SEQ, DM, 0, stream>>>(b_h, b_resid, norm_w + i * DM,
                                        b_hnbf, nullptr, i == 0);
    // in_proj GEMM: (4096x256)x(256x1024) -> xz (shared by both directions)
    int nInW = 2 * DI * DM;
    k_cvt<<<(nInW / 4 + 255) / 256, T256, 0, stream>>>(in_w + (size_t)i * nInW, w_in_bf, nInW);
    k_gemm_bf16<<<dim3(L_SEQ / 64, (2 * DI) / 64), T256, 0, stream>>>(
        b_hnbf, w_in_bf, b_xz, L_SEQ, 2 * DI, DM, 0);

    for (int dir = 0; dir < 2; ++dir) {
      size_t ld = (size_t)(i * 2 + dir);
      k_conv<<<(L_SEQ * DI) / T256, T256, 0, stream>>>(
          b_xz, conv_w + ld * DI * DC, conv_b + ld * DI, b_u[dir], b_ubf[dir], dir);
      int nXW = XDBW * DI;
      k_cvt<<<(nXW / 4 + 255) / 256, T256, 0, stream>>>(xp_w + ld * nXW, w_x_bf, nXW);
      k_gemm_bf16<<<dim3(L_SEQ / 64, (XDBW + 63) / 64), T256, 0, stream>>>(
          b_ubf[dir], w_x_bf, b_xdb[dir], L_SEQ, XDBW, DI, 0);
      k_dt<<<(L_SEQ * DI) / T256, T256, 0, stream>>>(
          b_xdb[dir], dt_w + ld * DI * DTR, dt_b + ld * DI, b_dt[dir]);
    }

    // both directions' scans run concurrently (gridDim.y = 2)
    k_scan<<<dim3(DI / 16, 2), T256, 0, stream>>>(
        b_u[0], b_u[1], b_dt[0], b_dt[1], b_xdb[0], b_xdb[1],
        A_log + (size_t)i * 2 * DI * DS, Dp + (size_t)i * 2 * DI);

    for (int dir = 0; dir < 2; ++dir)
      k_gate<<<(L_SEQ * DI) / T256, T256, 0, stream>>>(b_u[dir], b_xz, b_ubf[dir], dir);

    // out_proj GEMMs accumulate fwd + rev into layer output h
    int nOW = DM * DI;
    k_cvt<<<(nOW / 4 + 255) / 256, T256, 0, stream>>>(out_w + (size_t)i * nOW, w_out_bf, nOW);
    for (int dir = 0; dir < 2; ++dir)
      k_gemm_bf16<<<dim3(L_SEQ / 64, DM / 64), T256, 0, stream>>>(
          b_ubf[dir], w_out_bf, b_h, L_SEQ, DM, DI, dir);
  }

  // final: resid += h ; out = rmsnorm(resid, norm_f_w) in f32
  k_rmsnorm<<<L_SEQ, DM, 0, stream>>>(b_h, b_resid, normf_w, nullptr, out, 0);
}
